// IILoss_53798760349888
// MI455X (gfx1250) — matrix operand order repaired and checked by reading, not compiled, not run on hardware
//
#include <hip/hip_runtime.h>
#include <hip/hip_bf16.h>

// ---------------------------------------------------------------------------
// IILoss forward for MI455X (gfx1250, wave32).
//   k_init     : zero accumulators
//   k_segsum   : 1 streaming pass over embeddings; LDS-privatized segment sum
//                (260 KB LDS table per WG) + fused sum(e^2)  [bandwidth bound]
//   k_mu       : mu = sums/cnt, bf16 hi/lo split, sq norms (presence mask
//                baked in as 5e23), sum cnt*||mu||^2
//   k_gemm_min : mu@mu^T via v_wmma_f32_16x16x32_bf16 (bf16x3 emulated fp32),
//                branchless masked distance, wave min, global atomicMin
//   k_final    : intra = (sum e^2 - sum cnt*||mu||^2)/n ; inter = -min
// ---------------------------------------------------------------------------

#define NCLS   1000
#define NCLSP  1024
#define DIM    64
#define SUMSF  (NCLS * DIM)          // 64000 floats (sums)
#define TABF   (SUMSF + NCLS)        // 65000 floats (sums + counts)
#define MASKSQ 5.0e23f               // baked "absent class" norm -> dist >= 5e23
#define BIGF   1.0e24f

typedef __bf16 bf16;
typedef bf16  v16bf __attribute__((ext_vector_type(16)));
typedef bf16  v8bf  __attribute__((ext_vector_type(8)));
typedef float v8f   __attribute__((ext_vector_type(8)));

__device__ __forceinline__ unsigned mapF(float x) {
  unsigned u = __float_as_uint(x);
  return (u & 0x80000000u) ? ~u : (u | 0x80000000u);
}

__global__ void k_init(float* gsc, double* dacc, unsigned* minslot) {
  for (int i = blockIdx.x * blockDim.x + threadIdx.x; i < TABF;
       i += gridDim.x * blockDim.x)
    gsc[i] = 0.0f;
  if (blockIdx.x == 0 && threadIdx.x == 0) {
    dacc[0] = 0.0;
    dacc[1] = 0.0;
    *minslot = 0xFFFFFFFFu;
  }
}

// One pass over embeddings. Each wave's 32 lanes cover the 64 dims (as
// float2) of one row -> conflict-free ds_add_f32 into the privatized table.
__global__ void k_segsum(const float* __restrict__ emb,
                         const int* __restrict__ tgt,
                         float* __restrict__ gsc, double* dacc, int nrows) {
  extern __shared__ float ls[];                       // 65000 floats = 260 KB
  for (int i = threadIdx.x; i < TABF; i += blockDim.x) ls[i] = 0.0f;
  __syncthreads();

  float acc = 0.0f;                                   // sum of e^2 (partial)
  const int total = nrows * (DIM / 2);
  const float2* e2 = (const float2*)emb;
  for (int idx = blockIdx.x * blockDim.x + threadIdx.x; idx < total;
       idx += gridDim.x * blockDim.x) {
    int row  = idx >> 5;
    int pair = idx & 31;
    float2 v = e2[idx];
    int t = tgt[row];
    int b = t * DIM + pair * 2;
    atomicAdd(&ls[b],     v.x);                       // ds_add_f32
    atomicAdd(&ls[b + 1], v.y);
    if (pair == 0) atomicAdd(&ls[SUMSF + t], 1.0f);
    acc += v.x * v.x + v.y * v.y;
  }
  for (int off = 16; off; off >>= 1) acc += __shfl_xor(acc, off, 32);
  if ((threadIdx.x & 31) == 0) unsafeAtomicAdd(&dacc[0], (double)acc);

  __syncthreads();
  for (int i = threadIdx.x; i < TABF; i += blockDim.x) {
    float v = ls[i];
    if (v != 0.0f) unsafeAtomicAdd(&gsc[i], v);       // global_atomic_add_f32
  }
}

// One wave per class (padded to 1024). lane covers dims {lane, lane+32}.
// sq[c] carries the presence mask: absent/padded classes get MASKSQ so every
// distance touching them is automatically >= 5e23 (their mu is all-zero, so
// the Gram-matrix contribution is 0).
__global__ void k_mu(const float* __restrict__ gsc, bf16* __restrict__ muhi,
                     bf16* __restrict__ mulo, float* __restrict__ sq,
                     double* dacc) {
  int wv   = blockIdx.x * (blockDim.x >> 5) + (threadIdx.x >> 5);
  int lane = threadIdx.x & 31;
  if (wv >= NCLSP) return;
  float m0 = 0.0f, m1 = 0.0f, cnt = 0.0f;
  if (wv < NCLS) {
    cnt = gsc[SUMSF + wv];
    float inv = 1.0f / fmaxf(cnt, 1.0f);
    m0 = gsc[wv * DIM + lane]      * inv;
    m1 = gsc[wv * DIM + lane + 32] * inv;
  }
  bf16 h0 = (bf16)m0, h1 = (bf16)m1;
  bf16 l0 = (bf16)(m0 - (float)h0), l1 = (bf16)(m1 - (float)h1);
  muhi[wv * DIM + lane]      = h0;
  muhi[wv * DIM + lane + 32] = h1;
  mulo[wv * DIM + lane]      = l0;
  mulo[wv * DIM + lane + 32] = l1;
  float s = m0 * m0 + m1 * m1;
  for (int off = 16; off; off >>= 1) s += __shfl_xor(s, off, 32);
  if (lane == 0) {
    bool present = (wv < NCLS) && (cnt > 0.0f);
    sq[wv] = present ? s : MASKSQ;
    if (present) unsafeAtomicAdd(&dacc[1], (double)cnt * (double)s);
  }
}

// A-fragment (16x32 bf16): lane<16 holds row M=l, halves K[kb..kb+7] and
// K[kb+16..kb+23]; lanes 16..31 hold the +8 K offsets (ISA 7.12.2).
__device__ __forceinline__ v16bf loadA(const bf16* mat, int blk, int l, int hh,
                                       int kb) {
  const bf16* p = mat + (size_t)(blk * 16 + l) * DIM + kb + hh * 8;
  v8bf a0 = *(const v8bf*)p;
  v8bf a1 = *(const v8bf*)(p + 16);
  v16bf r;
#pragma unroll
  for (int e = 0; e < 8; ++e) { r[e] = a0[e]; r[8 + e] = a1[e]; }
  return r;
}

// B-fragment (32x16 bf16): lane<16 = column N=l with K[kb..kb+7] and
// K[kb+16..kb+23]; lanes 16..31 carry the +8 K offsets.
__device__ __forceinline__ v16bf loadB(const bf16* mat, int blk, int l, int hh,
                                       int kb) {
  const bf16* p = mat + (size_t)(blk * 16 + l) * DIM + kb + hh * 8;
  v8bf b0 = *(const v8bf*)p;
  v8bf b1 = *(const v8bf*)(p + 16);
  v16bf r;
#pragma unroll
  for (int e = 0; e < 8; ++e) { r[e] = b0[e]; r[8 + e] = b1[e]; }
  return r;
}

// One wave per 16x16 output tile of the 1024x1024 Gram matrix.
// bf16 hi/lo split: g = hi*hi + hi*lo + lo*hi  (~fp32 accuracy).
// Masking is branchless: sq[] already encodes absence; only the diagonal
// needs an explicit select.
__global__ void k_gemm_min(const bf16* __restrict__ muhi,
                           const bf16* __restrict__ mulo,
                           const float* __restrict__ sq,
                           unsigned* minslot) {
  int wv   = blockIdx.x * (blockDim.x >> 5) + (threadIdx.x >> 5);
  int bi   = wv >> 6;
  int bj   = wv & 63;
  int lane = threadIdx.x & 31;
  int l    = lane & 15;
  int hh   = lane >> 4;

  v8f c = {0.f, 0.f, 0.f, 0.f, 0.f, 0.f, 0.f, 0.f};
#pragma unroll
  for (int ks = 0; ks < 2; ++ks) {
    int kb = ks * 32;
    v16bf ahi = loadA(muhi, bi, l, hh, kb);
    v16bf alo = loadA(mulo, bi, l, hh, kb);
    v16bf bhi = loadB(muhi, bj, l, hh, kb);
    v16bf blo = loadB(mulo, bj, l, hh, kb);
    c = __builtin_amdgcn_wmma_f32_16x16x32_bf16(false, ahi, false, bhi,
                                                (short)0, c, false, false);
    c = __builtin_amdgcn_wmma_f32_16x16x32_bf16(false, ahi, false, blo,
                                                (short)0, c, false, false);
    c = __builtin_amdgcn_wmma_f32_16x16x32_bf16(false, alo, false, bhi,
                                                (short)0, c, false, false);
  }

  int j = bj * 16 + l;                     // this lane's column
  float sqj = sq[j];
  float best = BIGF;
#pragma unroll
  for (int v = 0; v < 8; ++v) {
    int i = bi * 16 + v + hh * 8;          // row for c[v] (C/D layout)
    float dd = sq[i] + sqj - 2.0f * c[v];
    best = fminf(best, (i != j) ? dd : BIGF);
  }
  for (int off = 16; off; off >>= 1)
    best = fminf(best, __shfl_xor(best, off, 32));
  if (lane == 0) atomicMin(minslot, mapF(best));
}

__global__ void k_final(const double* dacc, const unsigned* minslot,
                        float* out, int n) {
  if (blockIdx.x == 0 && threadIdx.x == 0) {
    double intra = (dacc[0] - dacc[1]) / (double)n;
    unsigned m = *minslot;
    unsigned u = (m & 0x80000000u) ? (m ^ 0x80000000u) : ~m;
    out[0] = (float)intra;
    out[1] = -__uint_as_float(u);
  }
}

extern "C" void kernel_launch(void* const* d_in, const int* in_sizes, int n_in,
                              void* d_out, int out_size, void* d_ws,
                              size_t ws_size, hipStream_t stream) {
  const float* emb = (const float*)d_in[0];
  const int*   tgt = (const int*)d_in[1];
  const int n = in_sizes[1];
  float* out = (float*)d_out;

  // Workspace layout (all offsets 16B-aligned):
  char* base = (char*)d_ws;
  float*    gsc     = (float*)base;                   //  65000 f32 (sums+cnts)
  bf16*     muhi    = (bf16*)(base + 260096);         //  1024*64 bf16
  bf16*     mulo    = (bf16*)(base + 391168);         //  1024*64 bf16
  float*    sq      = (float*)(base + 522240);        //  1024 f32
  double*   dacc    = (double*)(base + 526336);       //  [sum e^2, sum c*mu^2]
  unsigned* minslot = (unsigned*)(base + 526352);

  k_init<<<128, 256, 0, stream>>>(gsc, dacc, minslot);
  k_segsum<<<128, 256, TABF * sizeof(float), stream>>>(emb, tgt, gsc, dacc, n);
  k_mu<<<128, 256, 0, stream>>>(gsc, muhi, mulo, sq, dacc);
  k_gemm_min<<<512, 256, 0, stream>>>(muhi, mulo, sq, minslot);
  k_final<<<1, 32, 0, stream>>>(dacc, minslot, out, n);
}
